// CoTModel_83133386982050
// MI455X (gfx1250) — compile-verified, matrix operand order checked by reading, not run on hardware
//
#include <hip/hip_runtime.h>

#define Dm 1024
#define Hh 16
#define HDm 64
#define Ff 4096
#define Tt 1024
#define Bb 2
#define BT 2048
#define Vv 32000

typedef __attribute__((ext_vector_type(16))) __bf16 v16bf;
typedef __attribute__((ext_vector_type(8)))  __bf16 v8bf;
typedef __attribute__((ext_vector_type(8)))  float  v8f;
typedef __attribute__((ext_vector_type(4)))  float  v4f;

// Build a 16-element bf16 WMMA fragment from two contiguous 8-element (16B) loads.
// CDNA5 16-bit A/B operand layout: lane picks a row; K pattern is {0..7,16..23}
// for lanes 0-15 and {8..15,24..31} for lanes 16-31 (caller passes p0=row+lh*8,
// p1=row+16+lh*8).
static __device__ __forceinline__ v16bf frag16(const __bf16* p0, const __bf16* p1) {
  v8bf a = *(const v8bf*)p0;
  v8bf b = *(const v8bf*)p1;
  return __builtin_shufflevector(a, b, 0,1,2,3,4,5,6,7,8,9,10,11,12,13,14,15);
}

static __device__ __forceinline__ float gelu_f(float x) {
  float u = 0.7978845608028654f * (x + 0.044715f * x * x * x);
  return 0.5f * x * (1.0f + tanhf(u));
}

static __device__ __forceinline__ void gload16(float* dst, const float* src) {
  #pragma unroll
  for (int i = 0; i < 4; ++i) *(v4f*)&dst[4*i] = *(const v4f*)(src + 4*i);
}

static __device__ __forceinline__ void stage_cvt(__bf16* dst, const float* s16) {
  v8bf lo, hi;
  #pragma unroll
  for (int i = 0; i < 8; ++i) { lo[i] = (__bf16)s16[i]; hi[i] = (__bf16)s16[8+i]; }
  *(v8bf*)dst = lo; *(v8bf*)(dst + 8) = hi;
}

// C[M,N] = act(A[M,K] @ W[N,K]^T + bias[N]).  M,N multiples of 128; K multiple of 32.
// OUT: 0 = f32 row-major, 1 = bf16 row-major, 2 = bf16 col-major (out[col*ldo+row])
// ABF16: A is bf16 row-major; staged into LDS with global_load_async_to_lds_b128.
// Double-buffered LDS, 2-deep software pipeline, one barrier per K-step.
template<int OUT, bool GELU, bool ABF16>
__global__ __launch_bounds__(256) void gemm_wmma(
    const void* __restrict__ Ain, const float* __restrict__ W,
    const float* __restrict__ bias, void* __restrict__ Cout,
    int M, int N, int K, int ldo)
{
  __shared__ __attribute__((aligned(16))) __bf16 As[2][128][40];  // +8 pad, rows 16B aligned
  __shared__ __attribute__((aligned(16))) __bf16 Bs[2][128][40];
  const int tid  = threadIdx.x;
  const int lane = tid & 31, wid = tid >> 5;
  const int lr = lane & 15, lh = lane >> 4;
  const int bM = blockIdx.y * 128, bN = blockIdx.x * 128;
  const int wm = (wid & 3) * 32, wn = (wid >> 2) * 64;   // 4x2 wave grid, 32x64 per wave
  const int srow = tid >> 1, scol = (tid & 1) * 16;      // staging: half-row per thread

  const float*  Arow  = (const float*)Ain  + (size_t)(bM + srow) * K + scol;
  const __bf16* Abrow = (const __bf16*)Ain + (size_t)(bM + srow) * K + scol;
  const float*  Wrow  = W + (size_t)(bN + srow) * K + scol;

  v8f acc[2][4];
  #pragma unroll
  for (int mi = 0; mi < 2; ++mi)
    #pragma unroll
    for (int ni = 0; ni < 4; ++ni) acc[mi][ni] = {};

  alignas(16) float wreg[16], areg[16];

  // ---- prologue: stage tile 0 into buffer 0, start loads for tile 1 ----
  if constexpr (ABF16) {
    unsigned lds0 = (unsigned)(uintptr_t)&As[0][srow][scol];
    asm volatile("global_load_async_to_lds_b128 %0, %1, off\n\t"
                 "global_load_async_to_lds_b128 %2, %3, off"
                 :: "v"(lds0), "v"(Abrow), "v"(lds0 + 16u), "v"(Abrow + 8)
                 : "memory");
  } else {
    gload16(areg, Arow);
  }
  gload16(wreg, Wrow);
  if constexpr (!ABF16) stage_cvt(&As[0][srow][scol], areg);
  stage_cvt(&Bs[0][srow][scol], wreg);
  if (32 < K) {
    gload16(wreg, Wrow + 32);
    if constexpr (!ABF16) gload16(areg, Arow + 32);
  }
  if constexpr (ABF16) asm volatile("s_wait_asynccnt 0x0" ::: "memory");

  for (int kt = 0; kt < K; kt += 32) {
    __syncthreads();               // publish buf[cur]; retire reads of buf[cur^1]
    const int cur = (kt >> 5) & 1;
    // stage tile kt+32 into buf[cur^1] while buf[cur] feeds the matrix units
    if (kt + 32 < K) {
      if constexpr (ABF16) {
        const __bf16* ap = Abrow + kt + 32;
        unsigned lds = (unsigned)(uintptr_t)&As[cur ^ 1][srow][scol];
        asm volatile("global_load_async_to_lds_b128 %0, %1, off\n\t"
                     "global_load_async_to_lds_b128 %2, %3, off"
                     :: "v"(lds), "v"(ap), "v"(lds + 16u), "v"(ap + 8)
                     : "memory");
      } else {
        stage_cvt(&As[cur ^ 1][srow][scol], areg);
      }
      stage_cvt(&Bs[cur ^ 1][srow][scol], wreg);
    }
    // kick global loads for tile kt+64 (consumed next iteration)
    if (kt + 64 < K) {
      gload16(wreg, Wrow + kt + 64);
      if constexpr (!ABF16) gload16(areg, Arow + kt + 64);
    }
    if (kt + 96 < K) __builtin_prefetch(Wrow + kt + 96, 0, 0);  // global_prefetch_b8

    v16bf aF[2], bF[4];
    #pragma unroll
    for (int mi = 0; mi < 2; ++mi)
      aF[mi] = frag16(&As[cur][wm + mi*16 + lr][lh*8],
                      &As[cur][wm + mi*16 + lr][16 + lh*8]);
    #pragma unroll
    for (int ni = 0; ni < 4; ++ni)
      bF[ni] = frag16(&Bs[cur][wn + ni*16 + lr][lh*8],
                      &Bs[cur][wn + ni*16 + lr][16 + lh*8]);
    #pragma unroll
    for (int mi = 0; mi < 2; ++mi)
      #pragma unroll
      for (int ni = 0; ni < 4; ++ni)
        acc[mi][ni] = __builtin_amdgcn_wmma_f32_16x16x32_bf16(
            false, aF[mi], false, bF[ni], (short)0, acc[mi][ni], false, false);

    if (ABF16 && kt + 32 < K)       // own async copies done before next barrier
      asm volatile("s_wait_asynccnt 0x0" ::: "memory");
  }

  #pragma unroll
  for (int mi = 0; mi < 2; ++mi)
    #pragma unroll
    for (int ni = 0; ni < 4; ++ni) {
      const int col = bN + wn + ni*16 + lr;
      const float bv = bias[col];
      #pragma unroll
      for (int r = 0; r < 8; ++r) {
        const int row = bM + wm + mi*16 + lh*8 + r;   // C layout: lanes>=16 carry M+8
        float v = acc[mi][ni][r] + bv;
        if (GELU) v = gelu_f(v);
        if (OUT == 0)      ((float*)Cout)[(size_t)row * ldo + col] = v;
        else if (OUT == 1) ((__bf16*)Cout)[(size_t)row * ldo + col] = (__bf16)v;
        else               ((__bf16*)Cout)[(size_t)col * ldo + row] = (__bf16)v;
      }
    }
}

// Flash attention, fully on WMMA. Q,K: bf16 [B*T, D] row-major; Vt: bf16 [D, B*T].
// Grid (B*H, T/256), 8 waves/block, each wave owns 32 query rows independently.
__global__ __launch_bounds__(256) void flash_attn(
    const __bf16* __restrict__ Q, const __bf16* __restrict__ Kmat,
    const __bf16* __restrict__ Vt, float* __restrict__ O)
{
  __shared__ __attribute__((aligned(16))) __bf16 Pbuf[8][2][16*32];
  const int tid = threadIdx.x;
  const int lane = tid & 31, wid = tid >> 5;
  const int lr = lane & 15, lh = lane >> 4;
  const int b = blockIdx.x >> 4, h = blockIdx.x & 15;
  const int q0 = blockIdx.y * 256 + wid * 32;

  const __bf16* Qb = Q    + (size_t)b*Tt*Dm + (size_t)h*HDm;
  const __bf16* Kb = Kmat + (size_t)b*Tt*Dm + (size_t)h*HDm;

  v16bf aQ[2][2];
  #pragma unroll
  for (int mi = 0; mi < 2; ++mi) {
    const __bf16* qp = Qb + (size_t)(q0 + mi*16 + lr) * Dm;
    aQ[mi][0] = frag16(qp + lh*8,      qp + 16 + lh*8);
    aQ[mi][1] = frag16(qp + 32 + lh*8, qp + 48 + lh*8);
  }

  v8f o[2][4];
  #pragma unroll
  for (int mi = 0; mi < 2; ++mi)
    #pragma unroll
    for (int ni = 0; ni < 4; ++ni) o[mi][ni] = {};
  float mrun[2][8], lrun[2][8];
  #pragma unroll
  for (int mi = 0; mi < 2; ++mi)
    #pragma unroll
    for (int r = 0; r < 8; ++r) { mrun[mi][r] = -3.0e38f; lrun[mi][r] = 0.0f; }

  const int nkb = q0 / 32 + 1;   // causal: key blocks covering [0, q0+32)
  for (int jb = 0; jb < nkb; ++jb) {
    const int kbase = jb * 32;
    v16bf bK[2][2];
    #pragma unroll
    for (int nj = 0; nj < 2; ++nj) {
      const __bf16* kp = Kb + (size_t)(kbase + nj*16 + lr) * Dm;
      bK[nj][0] = frag16(kp + lh*8,      kp + 16 + lh*8);
      bK[nj][1] = frag16(kp + 32 + lh*8, kp + 48 + lh*8);
    }
    v8f s[2][2];
    #pragma unroll
    for (int mi = 0; mi < 2; ++mi)
      #pragma unroll
      for (int nj = 0; nj < 2; ++nj) {
        v8f z = {};
        z = __builtin_amdgcn_wmma_f32_16x16x32_bf16(false, aQ[mi][0], false, bK[nj][0], (short)0, z, false, false);
        z = __builtin_amdgcn_wmma_f32_16x16x32_bf16(false, aQ[mi][1], false, bK[nj][1], (short)0, z, false, false);
        s[mi][nj] = z;
      }
    // online softmax: each row's 16 columns live across a 16-lane half-wave
    #pragma unroll
    for (int mi = 0; mi < 2; ++mi) {
      #pragma unroll
      for (int r = 0; r < 8; ++r) {
        const int qrow = q0 + mi*16 + lh*8 + r;
        float s0 = s[mi][0][r] * 0.125f;            // 1/sqrt(64)
        float s1 = s[mi][1][r] * 0.125f;
        if (kbase + lr      > qrow) s0 = -1.0e30f;  // causal mask
        if (kbase + 16 + lr > qrow) s1 = -1.0e30f;
        float t = fmaxf(s0, s1);
        #pragma unroll
        for (int off = 1; off < 16; off <<= 1) t = fmaxf(t, __shfl_xor(t, off, 32));
        const float mnew = fmaxf(mrun[mi][r], t);
        const float resc = __expf(mrun[mi][r] - mnew);
        mrun[mi][r] = mnew;
        const float p0 = __expf(s0 - mnew);
        const float p1 = __expf(s1 - mnew);
        float ps = p0 + p1;
        #pragma unroll
        for (int off = 1; off < 16; off <<= 1) ps += __shfl_xor(ps, off, 32);
        lrun[mi][r] = lrun[mi][r] * resc + ps;
        #pragma unroll
        for (int ni = 0; ni < 4; ++ni) o[mi][ni][r] *= resc;
        Pbuf[wid][mi][(lh*8 + r)*32 + lr]      = (__bf16)p0;   // C-layout -> row-major LDS
        Pbuf[wid][mi][(lh*8 + r)*32 + 16 + lr] = (__bf16)p1;
      }
    }
    // LDS is in-order per wave; make the cross-lane re-read explicit
    asm volatile("s_wait_dscnt 0x0" ::: "memory");
    v16bf aP[2], bV[4];
    #pragma unroll
    for (int mi = 0; mi < 2; ++mi) {
      const __bf16* pp = &Pbuf[wid][mi][lr*32];
      aP[mi] = frag16(pp + lh*8, pp + 16 + lh*8);
    }
    #pragma unroll
    for (int ni = 0; ni < 4; ++ni) {
      const __bf16* vp = Vt + (size_t)(h*HDm + ni*16 + lr) * BT + (size_t)b*Tt + kbase;
      bV[ni] = frag16(vp + lh*8, vp + 16 + lh*8);
    }
    #pragma unroll
    for (int mi = 0; mi < 2; ++mi)
      #pragma unroll
      for (int ni = 0; ni < 4; ++ni)
        o[mi][ni] = __builtin_amdgcn_wmma_f32_16x16x32_bf16(
            false, aP[mi], false, bV[ni], (short)0, o[mi][ni], false, false);
  }

  #pragma unroll
  for (int mi = 0; mi < 2; ++mi)
    #pragma unroll
    for (int r = 0; r < 8; ++r) {
      const float inv = 1.0f / lrun[mi][r];
      const int row = q0 + mi*16 + lh*8 + r;
      #pragma unroll
      for (int ni = 0; ni < 4; ++ni) {
        const int col = h*HDm + ni*16 + lr;
        O[(size_t)(b*Tt + row) * Dm + col] = o[mi][ni][r] * inv;
      }
    }
}

// x = LN(x [+ addv]) * g + b ; one 256-thread block per row, D=1024.
template<bool RES>
__global__ __launch_bounds__(256) void layernorm_k(
    float* __restrict__ x, const float* __restrict__ addv,
    const float* __restrict__ g, const float* __restrict__ bta)
{
  __shared__ float red[2][8];
  const int row = blockIdx.x, tid = threadIdx.x;
  const int lane = tid & 31, wid = tid >> 5;
  float v[4];
  float s = 0.f, s2 = 0.f;
  #pragma unroll
  for (int i = 0; i < 4; ++i) {
    const int c = tid + i*256;
    float t = x[(size_t)row*Dm + c];
    if (RES) t += addv[(size_t)row*Dm + c];
    v[i] = t; s += t; s2 += t*t;
  }
  #pragma unroll
  for (int off = 16; off > 0; off >>= 1) {
    s += __shfl_xor(s, off, 32); s2 += __shfl_xor(s2, off, 32);
  }
  if (lane == 0) { red[0][wid] = s; red[1][wid] = s2; }
  __syncthreads();
  float ts = 0.f, ts2 = 0.f;
  #pragma unroll
  for (int w = 0; w < 8; ++w) { ts += red[0][w]; ts2 += red[1][w]; }
  const float mean = ts * (1.0f/Dm);
  const float var  = ts2 * (1.0f/Dm) - mean*mean;
  const float rs = rsqrtf(var + 1e-5f);
  #pragma unroll
  for (int i = 0; i < 4; ++i) {
    const int c = tid + i*256;
    x[(size_t)row*Dm + c] = (v[i] - mean) * rs * g[c] + bta[c];
  }
}

__global__ __launch_bounds__(256) void embed_k(
    const int* __restrict__ xi, const float* __restrict__ tokW,
    const float* __restrict__ pos, float* __restrict__ h)
{
  const int row = blockIdx.x;
  const int t = row & (Tt - 1);
  const int tok = xi[row];
  const int tid = threadIdx.x;
  #pragma unroll
  for (int i = 0; i < 4; ++i) {
    const int c = tid + i*256;
    h[(size_t)row*Dm + c] = tokW[(size_t)tok*Dm + c] + pos[(size_t)t*Dm + c];
  }
}

extern "C" void kernel_launch(void* const* d_in, const int* in_sizes, int n_in,
                              void* d_out, int out_size, void* d_ws, size_t ws_size,
                              hipStream_t stream) {
  (void)in_sizes; (void)n_in; (void)out_size; (void)ws_size;
  const int*   x     = (const int*)  d_in[0];
  const float* tok_W = (const float*)d_in[2];
  const float* pos   = (const float*)d_in[3];
  const float* Wq = (const float*)d_in[4];  const float* bq = (const float*)d_in[5];
  const float* Wk = (const float*)d_in[6];  const float* bk = (const float*)d_in[7];
  const float* Wv = (const float*)d_in[8];  const float* bv = (const float*)d_in[9];
  const float* Wo = (const float*)d_in[10]; const float* bo = (const float*)d_in[11];
  const float* ln1g = (const float*)d_in[12]; const float* ln1b = (const float*)d_in[13];
  const float* W1 = (const float*)d_in[14]; const float* b1 = (const float*)d_in[15];
  const float* W2 = (const float*)d_in[16]; const float* b2 = (const float*)d_in[17];
  const float* ln2g = (const float*)d_in[18]; const float* ln2b = (const float*)d_in[19];
  const float* mWq = (const float*)d_in[20]; const float* mbq = (const float*)d_in[21];
  const float* mWk = (const float*)d_in[22]; const float* mbk = (const float*)d_in[23];
  const float* mWv = (const float*)d_in[24]; const float* mbv = (const float*)d_in[25];
  const float* mWo = (const float*)d_in[26]; const float* mbo = (const float*)d_in[27];
  const float* mlng = (const float*)d_in[28]; const float* mlnb = (const float*)d_in[29];
  const float* tWq = (const float*)d_in[30]; const float* tbq = (const float*)d_in[31];
  const float* tWk = (const float*)d_in[32]; const float* tbk = (const float*)d_in[33];
  const float* tWv = (const float*)d_in[34]; const float* tbv = (const float*)d_in[35];
  const float* tWo = (const float*)d_in[36]; const float* tbo = (const float*)d_in[37];
  const float* tlng = (const float*)d_in[38]; const float* tlnb = (const float*)d_in[39];
  const float* ng = (const float*)d_in[40]; const float* nb = (const float*)d_in[41];
  const float* outW = (const float*)d_in[42]; const float* outb = (const float*)d_in[43];

  char* ws = (char*)d_ws;
  float*  h   = (float*)ws;  ws += (size_t)BT*Dm*4;   // hidden state
  float*  t0  = (float*)ws;  ws += (size_t)BT*Dm*4;   // attention output
  float*  t2  = (float*)ws;  ws += (size_t)BT*Dm*4;   // projection / FFN2 output
  __bf16* t1b = (__bf16*)ws; ws += (size_t)BT*Ff*2;   // FFN intermediate, bf16
  __bf16* qb  = (__bf16*)ws; ws += (size_t)BT*Dm*2;   // Q bf16 [BT,D]
  __bf16* kb  = (__bf16*)ws; ws += (size_t)BT*Dm*2;   // K bf16 [BT,D]
  __bf16* vt  = (__bf16*)ws; ws += (size_t)Dm*BT*2;   // V bf16 transposed [D,BT]

  dim3 blk(256);
  embed_k<<<BT, blk, 0, stream>>>(x, tok_W, pos, h);

  auto mha = [&](const float* wq, const float* bq_, const float* wk, const float* bk_,
                 const float* wv, const float* bv_, const float* wo, const float* bo_,
                 const float* g_, const float* be_) {
    dim3 gD(Dm/128, BT/128);
    gemm_wmma<1,false,false><<<gD, blk, 0, stream>>>(h, wq, bq_, qb, BT, Dm, Dm, Dm);
    gemm_wmma<1,false,false><<<gD, blk, 0, stream>>>(h, wk, bk_, kb, BT, Dm, Dm, Dm);
    gemm_wmma<2,false,false><<<gD, blk, 0, stream>>>(h, wv, bv_, vt, BT, Dm, Dm, BT);
    flash_attn<<<dim3(Bb*Hh, Tt/256), blk, 0, stream>>>(qb, kb, vt, t0);
    gemm_wmma<0,false,false><<<gD, blk, 0, stream>>>(t0, wo, bo_, t2, BT, Dm, Dm, Dm);
    layernorm_k<true><<<BT, blk, 0, stream>>>(h, t2, g_, be_);
  };

  for (int i = 0; i < 3; ++i) {
    mha(Wq + (size_t)i*Dm*Dm, bq + (size_t)i*Dm,
        Wk + (size_t)i*Dm*Dm, bk + (size_t)i*Dm,
        Wv + (size_t)i*Dm*Dm, bv + (size_t)i*Dm,
        Wo + (size_t)i*Dm*Dm, bo + (size_t)i*Dm,
        ln1g + (size_t)i*Dm, ln1b + (size_t)i*Dm);
    // FFN1: f32 in -> GELU -> bf16 out (feeds async-to-LDS FFN2)
    gemm_wmma<1,true ,false><<<dim3(Ff/128, BT/128), blk, 0, stream>>>(
        h, W1 + (size_t)i*Ff*Dm, b1 + (size_t)i*Ff, t1b, BT, Ff, Dm, Ff);
    // FFN2: bf16 A staged via global_load_async_to_lds_b128
    gemm_wmma<0,false,true ><<<dim3(Dm/128, BT/128), blk, 0, stream>>>(
        t1b, W2 + (size_t)i*Dm*Ff, b2 + (size_t)i*Dm, t2, BT, Dm, Ff, Dm);
    layernorm_k<true><<<BT, blk, 0, stream>>>(h, t2, ln2g + (size_t)i*Dm, ln2b + (size_t)i*Dm);
    if (i == 2)   // memory layer (i % 3 == 2, M = 1)
      mha(mWq, mbq, mWk, mbk, mWv, mbv, mWo, mbo, mlng, mlnb);
  }
  // thought attention
  mha(tWq, tbq, tWk, tbk, tWv, tbv, tWo, tbo, tlng, tlnb);
  // final norm
  layernorm_k<false><<<BT, blk, 0, stream>>>(h, nullptr, ng, nb);
  // LM head -> logits [B,T,V] f32
  gemm_wmma<0,false,false><<<dim3(Vv/128, BT/128), blk, 0, stream>>>(
      h, outW, outb, (float*)d_out, BT, Vv, Dm, Vv);
}
